// DimTransformerEncoderLayer_43989055046248
// MI455X (gfx1250) — matrix operand order repaired and checked by reading, not compiled, not run on hardware
//
#include <hip/hip_runtime.h>
#include <hip/hip_bf16.h>
#include <math.h>

// ---------------------------------------------------------------------------
// Types for CDNA5 WMMA (wave32). v_wmma_f32_16x16x32_bf16: A,B = v16bf, C/D = v8f.
// ---------------------------------------------------------------------------
typedef __bf16 bf16_t;
typedef __attribute__((ext_vector_type(16))) __bf16 v16bf;
typedef __attribute__((ext_vector_type(8)))  __bf16 v8bf;
typedef __attribute__((ext_vector_type(8)))  float  v8f;
typedef __attribute__((ext_vector_type(4)))  int    v4i;   // 16-byte copy unit

#define D_MODEL 1024
#define N_HEAD  16
#define HEAD_DIM 64
#define D_FF    4096
#define BATCH   4
#define SEQ     2048
#define ROWS    (BATCH * SEQ)   // 8192

// CDNA5 async global->LDS copy (ASYNCcnt path): 16B per lane, no VGPR staging.
// LDS address comes from the lds_off VGPR; global address is a 64-bit VGPR (no SGPR base).
__device__ __forceinline__ void async_copy16(const void* gaddr, void* lds_ptr) {
    const unsigned lds_off = (unsigned)(uintptr_t)lds_ptr;   // low 32 bits = LDS offset
    asm volatile("global_load_async_to_lds_b128 %0, %1, off"
                 :: "v"(lds_off), "v"(gaddr) : "memory");
}
__device__ __forceinline__ void wait_async_le4() { asm volatile("s_wait_asynccnt 0x4" ::: "memory"); }
__device__ __forceinline__ void wait_async_le2() { asm volatile("s_wait_asynccnt 0x2" ::: "memory"); }
__device__ __forceinline__ void wait_async_0()   { asm volatile("s_wait_asynccnt 0x0" ::: "memory"); }

// ---------------------------------------------------------------------------
// fp32 -> bf16 conversion (vectorized x4)
// ---------------------------------------------------------------------------
__global__ void cvt_f32_bf16(const float* __restrict__ in, bf16_t* __restrict__ out, int n4) {
    int i = blockIdx.x * blockDim.x + threadIdx.x;
    if (i < n4) {
        float4 v = ((const float4*)in)[i];
        out[4*i + 0] = (bf16_t)v.x;
        out[4*i + 1] = (bf16_t)v.y;
        out[4*i + 2] = (bf16_t)v.z;
        out[4*i + 3] = (bf16_t)v.w;
    }
}

// ---------------------------------------------------------------------------
// Tiled WMMA GEMM:  C[M,N] = A[M,K](bf16) * W[N,K]^T(bf16) + bias (+res)(+relu)
// Block: 256 thr (8 waves), tile 128x128. Wave: 32(M) x 64(N) = 2x4 WMMA tiles.
// K staged through double-buffered LDS slabs (2 x (128x32 A + 128x32 W), 32KB)
// filled with GLOBAL_LOAD_ASYNC_TO_LDS_B128 so tile t+1 fill overlaps tile t WMMAs.
// Fragments (A 16x32, B 32x16) use contiguous k: lane half h holds k 16h..16h+15 in A and B.
// ---------------------------------------------------------------------------
template<bool RELU, bool RES, bool OUT_BF16>
__global__ __launch_bounds__(256)
void gemm_bf16_wmma(const bf16_t* __restrict__ A, const bf16_t* __restrict__ W,
                    const float* __restrict__ bias, const float* __restrict__ res,
                    void* __restrict__ outp, int M, int N, int K) {
    __shared__ bf16_t As[2][128 * 32];
    __shared__ bf16_t Bs[2][128 * 32];

    const int tid  = threadIdx.x;
    const int lane = tid & 31;
    const int wave = tid >> 5;
    const int l15  = lane & 15;
    const int hi   = lane >> 4;

    const int bm = blockIdx.y * 128;
    const int bn = blockIdx.x * 128;
    const int wm = (wave & 3) * 32;   // wave row offset in block tile
    const int wn = (wave >> 2) * 64;  // wave col offset in block tile

    // This thread's 2 chunks (16B each) of the A tile and of the W tile.
    const int c0 = tid, c1 = tid + 256;            // chunk ids 0..511
    const int r0 = c0 >> 2, o0 = (c0 & 3) * 8;     // row 0..127, half offset 0..24
    const int r1 = c1 >> 2, o1 = (c1 & 3) * 8;

    // Issue the 4 async fills (2 for A, 2 for W) of buffer `buf` at K offset k0.
    auto issue_tile = [&](int buf, int k0) {
        async_copy16(&A[(size_t)(bm + r0) * K + k0 + o0], &As[buf][r0 * 32 + o0]);
        async_copy16(&A[(size_t)(bm + r1) * K + k0 + o1], &As[buf][r1 * 32 + o1]);
        async_copy16(&W[(size_t)(bn + r0) * K + k0 + o0], &Bs[buf][r0 * 32 + o0]);
        async_copy16(&W[(size_t)(bn + r1) * K + k0 + o1], &Bs[buf][r1 * 32 + o1]);
    };

    v8f acc[2][4];
    #pragma unroll
    for (int i = 0; i < 2; ++i)
        #pragma unroll
        for (int j = 0; j < 4; ++j)
            #pragma unroll
            for (int e = 0; e < 8; ++e) acc[i][j][e] = 0.0f;

    issue_tile(0, 0);   // prologue prefetch

    int buf = 0;
    for (int k0 = 0; k0 < K; k0 += 32, buf ^= 1) {
        if (k0 + 32 < K) {
            issue_tile(buf ^ 1, k0 + 32);   // overlap next fill with this tile's math
            wait_async_le4();               // oldest 4 (current tile) are complete
        } else {
            wait_async_0();
        }
        __syncthreads();   // every wave waited its own ASYNCcnt -> all LDS writes visible

        // A fragments (rows wm, wm+16): lane<16 holds K 0..15, lane>=16 holds K 16..31 (contiguous)
        v16bf afrag[2];
        #pragma unroll
        for (int i = 0; i < 2; ++i) {
            const bf16_t* p = &As[buf][(wm + i * 16 + l15) * 32 + hi * 16];
            v8bf lo = *(const v8bf*)p;
            v8bf hh = *(const v8bf*)(p + 8);
            #pragma unroll
            for (int e = 0; e < 8; ++e) { afrag[i][e] = lo[e]; afrag[i][8 + e] = hh[e]; }
        }
        // B fragments (cols wn + j*16): lane<16 holds K 0..15, lane>=16 holds K 16..31 (contiguous)
        v16bf bfrag[4];
        #pragma unroll
        for (int j = 0; j < 4; ++j)
            bfrag[j] = *(const v16bf*)&Bs[buf][(wn + j * 16 + l15) * 32 + hi * 16];

        #pragma unroll
        for (int i = 0; i < 2; ++i)
            #pragma unroll
            for (int j = 0; j < 4; ++j)
                acc[i][j] = __builtin_amdgcn_wmma_f32_16x16x32_bf16(
                    false, afrag[i], false, bfrag[j], (short)0, acc[i][j], false, false);

        __syncthreads();   // all waves done reading `buf` before it is refilled
    }

    // Epilogue. C/D layout: VGPR r -> row r + 8*hi, col = lane&15.
    #pragma unroll
    for (int i = 0; i < 2; ++i) {
        #pragma unroll
        for (int j = 0; j < 4; ++j) {
            const int col = bn + wn + j * 16 + l15;
            const float bv = bias[col];
            #pragma unroll
            for (int r = 0; r < 8; ++r) {
                const int row = bm + wm + i * 16 + r + hi * 8;
                float v = acc[i][j][r] + bv;
                if (RES)  v += res[(size_t)row * N + col];
                if (RELU) v = v > 0.0f ? v : 0.0f;
                if (OUT_BF16) ((bf16_t*)outp)[(size_t)row * N + col] = (bf16_t)v;
                else          ((float*)outp)[(size_t)row * N + col]  = v;
            }
        }
    }
}

// ---------------------------------------------------------------------------
// Per-(b,h) data-dependent scale: inv_scale = rsqrt(mean(K^2 over S x hd) + eps)
// ---------------------------------------------------------------------------
__global__ __launch_bounds__(256)
void kscale_kernel(const bf16_t* __restrict__ Kb, float* __restrict__ inv_scale) {
    const int bh = blockIdx.x;                // 0..63
    const int b = bh >> 4, h = bh & 15;
    const size_t base = (size_t)b * SEQ * D_MODEL + h * HEAD_DIM;
    float s = 0.0f;
    for (int i = threadIdx.x; i < SEQ * HEAD_DIM; i += 256) {
        const int sr = i >> 6, d = i & 63;
        const float v = (float)Kb[base + (size_t)sr * D_MODEL + d];
        s += v * v;
    }
    __shared__ float red[256];
    red[threadIdx.x] = s;
    __syncthreads();
    for (int off = 128; off > 0; off >>= 1) {
        if (threadIdx.x < off) red[threadIdx.x] += red[threadIdx.x + off];
        __syncthreads();
    }
    if (threadIdx.x == 0)
        inv_scale[bh] = rsqrtf(red[0] * (1.0f / (SEQ * HEAD_DIM)) + 1e-8f);
}

// ---------------------------------------------------------------------------
// Flash attention (non-causal, data-dependent scale).
// Grid: (S/128, B*H). Block: 256 thr = 8 waves, wave w handles 16 query rows.
// K/V tiles (32 keys) double-buffered in LDS via async global->LDS copies:
// tile t+1 fill overlaps tile t's WMMAs + softmax VALU work. Online softmax fp32.
// P is re-laid out (C tile -> A fragment) through a per-wave LDS buffer with an
// explicit CDNA5 s_wait_dscnt for wave-local store->load ordering.
// ---------------------------------------------------------------------------
__global__ __launch_bounds__(256)
void flash_attn(const bf16_t* __restrict__ Q, const bf16_t* __restrict__ Kb,
                const bf16_t* __restrict__ Vb, const float* __restrict__ inv_scale,
                bf16_t* __restrict__ O) {
    __shared__ bf16_t Kt[2][32 * 64];
    __shared__ bf16_t Vt[2][32 * 64];
    __shared__ bf16_t Pb[8][16 * 32];

    const int bh = blockIdx.y;
    const int b = bh >> 4, h = bh & 15;
    const int lane = threadIdx.x & 31, wave = threadIdx.x >> 5;
    const int l15 = lane & 15, hi = lane >> 4;
    const int qw = blockIdx.x * 128 + wave * 16;   // wave's 16 query rows

    const float isc = inv_scale[bh];

    // This thread's 16B chunk of the 32x64 K/V tiles.
    const int key_ld = threadIdx.x >> 3;
    const int o_ld   = (threadIdx.x & 7) * 8;
    auto issue_kv = [&](int buf, int kt) {
        const size_t g = ((size_t)b * SEQ + kt * 32 + key_ld) * D_MODEL + h * HEAD_DIM + o_ld;
        async_copy16(&Kb[g], &Kt[buf][key_ld * 64 + o_ld]);
        async_copy16(&Vb[g], &Vt[buf][key_ld * 64 + o_ld]);
    };

    // Preload Q fragments: 16 rows x 64 dims -> 2 A-fragments (K=32 each)
    v16bf qf[2];
    {
        const size_t qrow = ((size_t)b * SEQ + qw + l15) * D_MODEL + h * HEAD_DIM;
        #pragma unroll
        for (int c = 0; c < 2; ++c) {
            const bf16_t* p = &Q[qrow + c * 32 + hi * 16];
            v8bf lo = *(const v8bf*)p;
            v8bf hh = *(const v8bf*)(p + 8);
            #pragma unroll
            for (int e = 0; e < 8; ++e) { qf[c][e] = lo[e]; qf[c][8 + e] = hh[e]; }
        }
    }

    v8f o_acc[4];
    #pragma unroll
    for (int j = 0; j < 4; ++j)
        #pragma unroll
        for (int e = 0; e < 8; ++e) o_acc[j][e] = 0.0f;
    float m_run[8], l_run[8];
    #pragma unroll
    for (int r = 0; r < 8; ++r) { m_run[r] = -3.0e38f; l_run[r] = 0.0f; }

    issue_kv(0, 0);   // prologue prefetch

    int buf = 0;
    for (int kt = 0; kt < SEQ / 32; ++kt, buf ^= 1) {
        if (kt + 1 < SEQ / 32) {
            issue_kv(buf ^ 1, kt + 1);
            wait_async_le2();           // oldest 2 (current K/V tile) complete
        } else {
            wait_async_0();
        }
        __syncthreads();

        // Scores 16x32 = two 16x16 WMMA tiles, K-dim chained over hd=64 (2x32)
        v8f sc[2];
        #pragma unroll
        for (int j = 0; j < 2; ++j) {
            #pragma unroll
            for (int e = 0; e < 8; ++e) sc[j][e] = 0.0f;
            #pragma unroll
            for (int c = 0; c < 2; ++c) {
                v16bf kf = *(const v16bf*)&Kt[buf][(j * 16 + l15) * 64 + c * 32 + hi * 16];
                sc[j] = __builtin_amdgcn_wmma_f32_16x16x32_bf16(
                    false, qf[c], false, kf, (short)0, sc[j], false, false);
            }
        }

        // Online softmax: row reductions across the 16-lane half (wave32)
        float mnew[8], alpha[8];
        #pragma unroll
        for (int r = 0; r < 8; ++r) {
            float s0 = sc[0][r] * isc, s1 = sc[1][r] * isc;
            sc[0][r] = s0; sc[1][r] = s1;
            float mx = fmaxf(s0, s1);
            mx = fmaxf(mx, __shfl_xor(mx, 1, 32));
            mx = fmaxf(mx, __shfl_xor(mx, 2, 32));
            mx = fmaxf(mx, __shfl_xor(mx, 4, 32));
            mx = fmaxf(mx, __shfl_xor(mx, 8, 32));
            mnew[r]  = fmaxf(m_run[r], mx);
            alpha[r] = __expf(m_run[r] - mnew[r]);
            m_run[r] = mnew[r];
        }
        #pragma unroll
        for (int r = 0; r < 8; ++r) {
            const float p0 = __expf(sc[0][r] - mnew[r]);
            const float p1 = __expf(sc[1][r] - mnew[r]);
            float rs = p0 + p1;
            rs += __shfl_xor(rs, 1, 32);
            rs += __shfl_xor(rs, 2, 32);
            rs += __shfl_xor(rs, 4, 32);
            rs += __shfl_xor(rs, 8, 32);
            l_run[r] = l_run[r] * alpha[r] + rs;
            const int prow = r + hi * 8;
            Pb[wave][prow * 32 + l15]      = (bf16_t)p0;
            Pb[wave][prow * 32 + 16 + l15] = (bf16_t)p1;
        }
        #pragma unroll
        for (int j = 0; j < 4; ++j)
            #pragma unroll
            for (int r = 0; r < 8; ++r) o_acc[j][r] *= alpha[r];

        // Wave-local LDS ordering: P stores must land before A-fragment reload.
        asm volatile("s_wait_dscnt 0x0" ::: "memory");

        v16bf pf;
        {
            const bf16_t* p = &Pb[wave][l15 * 32 + hi * 16];
            v8bf lo = *(const v8bf*)p;
            v8bf hh = *(const v8bf*)(p + 8);
            #pragma unroll
            for (int e = 0; e < 8; ++e) { pf[e] = lo[e]; pf[8 + e] = hh[e]; }
        }
        // O += P(16x32) x V(32x64): 4 d-tiles. B frag gathered column-wise from LDS.
        #pragma unroll
        for (int j = 0; j < 4; ++j) {
            v16bf vf;
            #pragma unroll
            for (int e = 0; e < 16; ++e)
                vf[e] = Vt[buf][(hi * 16 + e) * 64 + j * 16 + l15];
            o_acc[j] = __builtin_amdgcn_wmma_f32_16x16x32_bf16(
                false, pf, false, vf, (short)0, o_acc[j], false, false);
        }

        __syncthreads();   // all waves done reading `buf` before it is refilled
    }

    // Normalize and store O in (B,S,D) bf16 layout for the output projection GEMM.
    #pragma unroll
    for (int j = 0; j < 4; ++j)
        #pragma unroll
        for (int r = 0; r < 8; ++r) {
            const size_t row = (size_t)b * SEQ + qw + r + hi * 8;
            O[row * D_MODEL + h * HEAD_DIM + j * 16 + l15] = (bf16_t)(o_acc[j][r] / l_run[r]);
        }
}

// ---------------------------------------------------------------------------
// Row LayerNorm (block per row, two-pass mean/var like the reference)
// ---------------------------------------------------------------------------
__device__ __forceinline__ float block_sum(float v) {
    __shared__ float red[256];
    red[threadIdx.x] = v;
    __syncthreads();
    for (int off = 128; off > 0; off >>= 1) {
        if (threadIdx.x < off) red[threadIdx.x] += red[threadIdx.x + off];
        __syncthreads();
    }
    const float r = red[0];
    __syncthreads();
    return r;
}

template<bool WRITE_BF16>
__global__ __launch_bounds__(256)
void layernorm_kernel(const float* __restrict__ X, const float* __restrict__ g,
                      const float* __restrict__ be, float* __restrict__ Y,
                      bf16_t* __restrict__ Yb) {
    const size_t row = blockIdx.x;
    const float* xr = X + row * D_MODEL;
    float v[4], s = 0.0f;
    #pragma unroll
    for (int i = 0; i < 4; ++i) { v[i] = xr[threadIdx.x + i * 256]; s += v[i]; }
    const float mu = block_sum(s) * (1.0f / D_MODEL);
    float d2 = 0.0f;
    #pragma unroll
    for (int i = 0; i < 4; ++i) { const float d = v[i] - mu; d2 += d * d; }
    const float var = block_sum(d2) * (1.0f / D_MODEL);
    const float rstd = rsqrtf(var + 1e-5f);
    #pragma unroll
    for (int i = 0; i < 4; ++i) {
        const int c = threadIdx.x + i * 256;
        const float y = (v[i] - mu) * rstd * g[c] + be[c];
        Y[row * D_MODEL + c] = y;
        if (WRITE_BF16) Yb[row * D_MODEL + c] = (bf16_t)y;
    }
}

// ---------------------------------------------------------------------------
// Host orchestration
// ---------------------------------------------------------------------------
extern "C" void kernel_launch(void* const* d_in, const int* in_sizes, int n_in,
                              void* d_out, int out_size, void* d_ws, size_t ws_size,
                              hipStream_t stream) {
    const float* x   = (const float*)d_in[0];
    const float* wq  = (const float*)d_in[1];  const float* bq  = (const float*)d_in[2];
    const float* wk  = (const float*)d_in[3];  const float* bk  = (const float*)d_in[4];
    const float* wv  = (const float*)d_in[5];  const float* bv  = (const float*)d_in[6];
    const float* wo  = (const float*)d_in[7];  const float* bo  = (const float*)d_in[8];
    const float* w1  = (const float*)d_in[9];  const float* b1  = (const float*)d_in[10];
    const float* w2  = (const float*)d_in[11]; const float* b2  = (const float*)d_in[12];
    const float* g1  = (const float*)d_in[13]; const float* be1 = (const float*)d_in[14];
    const float* g2  = (const float*)d_in[15]; const float* be2 = (const float*)d_in[16];

    // Workspace bump allocator (~242 MB total).
    char* ws = (char*)d_ws;
    size_t off = 0;
    auto alloc = [&](size_t bytes) -> char* {
        char* p = ws + off;
        off += (bytes + 255) & ~(size_t)255;
        return p;
    };
    bf16_t* xb   = (bf16_t*)alloc((size_t)ROWS * D_MODEL * 2);      // bf16 x (reused as attn out)
    bf16_t* wqb  = (bf16_t*)alloc((size_t)D_MODEL * D_MODEL * 2);
    bf16_t* wkb  = (bf16_t*)alloc((size_t)D_MODEL * D_MODEL * 2);
    bf16_t* wvb  = (bf16_t*)alloc((size_t)D_MODEL * D_MODEL * 2);
    bf16_t* wob  = (bf16_t*)alloc((size_t)D_MODEL * D_MODEL * 2);
    bf16_t* w1b  = (bf16_t*)alloc((size_t)D_FF * D_MODEL * 2);
    bf16_t* w2b  = (bf16_t*)alloc((size_t)D_MODEL * D_FF * 2);
    bf16_t* qb   = (bf16_t*)alloc((size_t)ROWS * D_MODEL * 2);
    bf16_t* kb   = (bf16_t*)alloc((size_t)ROWS * D_MODEL * 2);
    bf16_t* vbuf = (bf16_t*)alloc((size_t)ROWS * D_MODEL * 2);
    float*  h1   = (float*)alloc((size_t)ROWS * D_MODEL * 4);       // wo-gemm out, reused as ffn2 out
    float*  y1f  = (float*)alloc((size_t)ROWS * D_MODEL * 4);
    bf16_t* y1b  = (bf16_t*)alloc((size_t)ROWS * D_MODEL * 2);
    bf16_t* ffb  = (bf16_t*)alloc((size_t)ROWS * D_FF * 2);
    float*  iscl = (float*)alloc(256);
    (void)ws_size; (void)in_sizes; (void)n_in; (void)out_size;

    auto cvt = [&](const float* src, bf16_t* dst, size_t n) {
        const int n4 = (int)(n / 4);
        cvt_f32_bf16<<<(n4 + 255) / 256, 256, 0, stream>>>(src, dst, n4);
    };
    cvt(x,  xb,  (size_t)ROWS * D_MODEL);
    cvt(wq, wqb, (size_t)D_MODEL * D_MODEL);
    cvt(wk, wkb, (size_t)D_MODEL * D_MODEL);
    cvt(wv, wvb, (size_t)D_MODEL * D_MODEL);
    cvt(wo, wob, (size_t)D_MODEL * D_MODEL);
    cvt(w1, w1b, (size_t)D_FF * D_MODEL);
    cvt(w2, w2b, (size_t)D_MODEL * D_FF);

    const dim3 blk(256);
    // QKV projections: (8192 x 1024) x (1024 x 1024)^T, bf16 out
    {
        const dim3 grid(D_MODEL / 128, ROWS / 128);
        gemm_bf16_wmma<false, false, true><<<grid, blk, 0, stream>>>(xb, wqb, bq, nullptr, qb,   ROWS, D_MODEL, D_MODEL);
        gemm_bf16_wmma<false, false, true><<<grid, blk, 0, stream>>>(xb, wkb, bk, nullptr, kb,   ROWS, D_MODEL, D_MODEL);
        gemm_bf16_wmma<false, false, true><<<grid, blk, 0, stream>>>(xb, wvb, bv, nullptr, vbuf, ROWS, D_MODEL, D_MODEL);
    }
    // Data-dependent scale per (b,h)
    kscale_kernel<<<BATCH * N_HEAD, blk, 0, stream>>>(kb, iscl);
    // Flash attention -> bf16 attention output (reuses xb, which is dead now)
    bf16_t* attnb = xb;
    flash_attn<<<dim3(SEQ / 128, BATCH * N_HEAD), blk, 0, stream>>>(qb, kb, vbuf, iscl, attnb);
    // Output projection + residual(x): fp32 out
    gemm_bf16_wmma<false, true, false><<<dim3(D_MODEL / 128, ROWS / 128), blk, 0, stream>>>(
        attnb, wob, bo, x, h1, ROWS, D_MODEL, D_MODEL);
    // LayerNorm 1 -> y1 (fp32 for residual, bf16 for FFN input)
    layernorm_kernel<true><<<ROWS, blk, 0, stream>>>(h1, g1, be1, y1f, y1b);
    // FFN1 + ReLU: (8192 x 1024) x (4096 x 1024)^T -> bf16
    gemm_bf16_wmma<true, false, true><<<dim3(D_FF / 128, ROWS / 128), blk, 0, stream>>>(
        y1b, w1b, b1, nullptr, ffb, ROWS, D_FF, D_MODEL);
    // FFN2 + residual(y1): (8192 x 4096) x (1024 x 4096)^T -> fp32 (reuse h1)
    gemm_bf16_wmma<false, true, false><<<dim3(D_MODEL / 128, ROWS / 128), blk, 0, stream>>>(
        ffb, w2b, b2, y1f, h1, ROWS, D_MODEL, D_FF);
    // LayerNorm 2 -> d_out (fp32)
    layernorm_kernel<false><<<ROWS, blk, 0, stream>>>(h1, g2, be2, (float*)d_out, nullptr);
}
